// GCGRU_59493886984732
// MI455X (gfx1250) — matrix operand order repaired and checked by reading, not compiled, not run on hardware
//
#include <hip/hip_runtime.h>
#include <hip/hip_bf16.h>

#define NNODE 2048
#define BATCH 16
#define CIN   66      // Ci + H = 2 + 64
#define KI    198     // cheb_k * CIN
#define HID   64
#define OG    128     // gate output (2*H)
#define OU    64      // update output
#define DDIM  10
#define FDIM  1056    // BATCH * CIN

typedef __attribute__((ext_vector_type(16))) _Float16 v16h;
typedef __attribute__((ext_vector_type(8)))  _Float16 v8h;
typedef __attribute__((ext_vector_type(8)))  float    v8f;

// ---------------------------------------------------------------------------
// WMMA helpers: CDNA5 v_wmma_f32_16x16x32_f16 (wave32), per ISA §7.12.2 layouts
// ---------------------------------------------------------------------------
__device__ __forceinline__ v8f wmma_f16(v16h a, v16h b, v8f c) {
  // 8 args: (neg_a, A, neg_b, B, c_mod, C, reuse_a, reuse_b)
  return __builtin_amdgcn_wmma_f32_16x16x32_f16(false, a, false, b, (short)0, c,
                                                false, false);
}

// A-matrix 16x32 f16 fragment: lane (m = lane%16, half = lane/16) holds
// K = half*8 + 0..7 (vgpr0-3) and K = 16 + half*8 + 0..7 (vgpr4-7).
// p points at rowBase + kk + half*8 in a K-contiguous row.
__device__ __forceinline__ v16h load_frag_a(const _Float16* p) {
  v8h lo = *(const v8h*)(p);
  v8h hi = *(const v8h*)(p + 16);
  v16h r;
#pragma unroll
  for (int i = 0; i < 8; ++i) { r[i] = lo[i]; r[i + 8] = hi[i]; }
  return r;
}

// B-matrix 32x16 f16 fragment: lane (n = lane%16, half = lane/16) holds
// K = half*8 + 0..7 and K = 16 + half*8 + 0..7. p points at rowBase + kk + half*8.
__device__ __forceinline__ v16h load_frag_b(const _Float16* p) {
  v8h lo = *(const v8h*)(p);
  v8h hi = *(const v8h*)(p + 16);
  v16h r;
#pragma unroll
  for (int i = 0; i < 8; ++i) { r[i] = lo[i]; r[i + 8] = hi[i]; }
  return r;
}

// ---------------------------------------------------------------------------
// 1) A = softmax(relu(E @ E^T), axis=1), stored f16 row-major [N, N]
// ---------------------------------------------------------------------------
__global__ __launch_bounds__(256) void build_A_kernel(const float* __restrict__ E,
                                                      _Float16* __restrict__ A16) {
  __shared__ float row[NNODE];
  __shared__ float red[256];
  __shared__ float Ei[DDIM];
  const int i = blockIdx.x;
  const int t = threadIdx.x;
  if (t < DDIM) Ei[t] = E[i * DDIM + t];
  __syncthreads();

  float lmax = -1e30f;
  for (int j = t; j < NNODE; j += 256) {
    float d = 0.f;
#pragma unroll
    for (int dd = 0; dd < DDIM; ++dd) d += Ei[dd] * E[j * DDIM + dd];
    d = d > 0.f ? d : 0.f;
    row[j] = d;
    lmax = fmaxf(lmax, d);
  }
  red[t] = lmax;
  __syncthreads();
  for (int s = 128; s > 0; s >>= 1) {
    if (t < s) red[t] = fmaxf(red[t], red[t + s]);
    __syncthreads();
  }
  const float mx = red[0];
  __syncthreads();

  float lsum = 0.f;
  for (int j = t; j < NNODE; j += 256) {
    float e = expf(row[j] - mx);
    row[j] = e;
    lsum += e;
  }
  red[t] = lsum;
  __syncthreads();
  for (int s = 128; s > 0; s >>= 1) {
    if (t < s) red[t] += red[t + s];
    __syncthreads();
  }
  const float inv = 1.f / red[0];
  for (int j = t; j < NNODE; j += 256)
    A16[(size_t)i * NNODE + j] = (_Float16)(row[j] * inv);
}

// ---------------------------------------------------------------------------
// 2) Pack inp = concat(x, state) into transposed f16 feature matrix
//    X0T[f, n], f = b*66 + c
// ---------------------------------------------------------------------------
__global__ __launch_bounds__(256) void pack_inp_kernel(const float* __restrict__ x,
                                                       const float* __restrict__ state,
                                                       _Float16* __restrict__ X0T) {
  const int idx = blockIdx.x * 256 + threadIdx.x;
  if (idx >= FDIM * NNODE) return;
  const int f = idx / NNODE;
  const int n = idx % NNODE;
  const int b = f / CIN;
  const int c = f % CIN;
  const float v = (c < 2) ? x[((size_t)b * NNODE + n) * 2 + c]
                          : state[((size_t)b * NNODE + n) * HID + (c - 2)];
  X0T[idx] = (_Float16)v;
}

// ---------------------------------------------------------------------------
// 3) Graph-propagation GEMM: OutT = alpha * (A16 @ B) + beta * Prev, all f16
//    stored transposed [FDIM, NNODE]. A16 is [M=2048, K=2048] f16 row-major.
//    Block: 4 waves; each wave computes 64(M) x 32(F) with 8 accumulators,
//    reusing each A fragment across 2 F-subtiles (12 b128 loads per 8 wmma).
//    Grid: (FDIM/32, NNODE/256) = (33, 8).
// ---------------------------------------------------------------------------
__global__ __launch_bounds__(128) void gemm_prop_kernel(
    const _Float16* __restrict__ A16, const _Float16* __restrict__ BT,
    const _Float16* __restrict__ PrevT, _Float16* __restrict__ OutT,
    float alpha, float beta) {
  const int t = threadIdx.x;
  const int wave = t >> 5;
  const int lane = t & 31;
  const int half = lane >> 4;
  const int lp = lane & 15;

  const int f0 = blockIdx.x * 32;              // two 16-wide F subtiles
  const int m0 = blockIdx.y * 256 + wave * 64; // four 16-row M subtiles

  v8f acc[4][2];
#pragma unroll
  for (int s = 0; s < 4; ++s)
#pragma unroll
    for (int j = 0; j < 2; ++j)
      acc[s][j] = (v8f){0.f, 0.f, 0.f, 0.f, 0.f, 0.f, 0.f, 0.f};

  const _Float16* brow0 = BT + (size_t)(f0 + lp) * NNODE;
  const _Float16* brow1 = BT + (size_t)(f0 + 16 + lp) * NNODE;
  for (int kk = 0; kk < NNODE; kk += 32) {
    const v16h bf0 = load_frag_b(brow0 + kk + half * 8);
    const v16h bf1 = load_frag_b(brow1 + kk + half * 8);
#pragma unroll
    for (int s = 0; s < 4; ++s) {
      const v16h af =
          load_frag_a(A16 + (size_t)(m0 + s * 16 + lp) * NNODE + kk + half * 8);
      acc[s][0] = wmma_f16(af, bf0, acc[s][0]);
      acc[s][1] = wmma_f16(af, bf1, acc[s][1]);
    }
  }

  // Epilogue: D element r = C[m0 + s*16 + half*8 + r, f]; 8 contiguous cols
  // in the transposed output row f -> one b128 f16 store per (s, j).
#pragma unroll
  for (int j = 0; j < 2; ++j) {
    const int f = f0 + j * 16 + lp;
#pragma unroll
    for (int s = 0; s < 4; ++s) {
      const int colbase = m0 + s * 16 + half * 8;
      v8h ov;
      if (beta != 0.f) {
        const v8h pv = *(const v8h*)(PrevT + (size_t)f * NNODE + colbase);
#pragma unroll
        for (int r = 0; r < 8; ++r)
          ov[r] = (_Float16)(alpha * acc[s][j][r] + beta * (float)pv[r]);
      } else {
#pragma unroll
        for (int r = 0; r < 8; ++r) ov[r] = (_Float16)(alpha * acc[s][j][r]);
      }
      *(v8h*)(OutT + (size_t)f * NNODE + colbase) = ov;
    }
  }
}

// ---------------------------------------------------------------------------
// 4) Gate per-node kernel: z_r = sigmoid(sum_ki x_g * W_n + b_n)
//    W_n[ki,o] = sum_d E[n,d] * pool[d,ki,o]. Writes z to zbuf and overwrites
//    the state rows (c>=2) of X0T column n with f16(r * state) -> candidate X.
// ---------------------------------------------------------------------------
__global__ __launch_bounds__(256) void gate_node_kernel(
    const _Float16* __restrict__ X0T, const _Float16* __restrict__ X1T,
    const _Float16* __restrict__ X2T, const float* __restrict__ E,
    const float* __restrict__ Wpool, const float* __restrict__ Bpool,
    const float* __restrict__ state, float* __restrict__ zbuf,
    _Float16* __restrict__ XcT) {
  __shared__ float    En[DDIM];
  __shared__ _Float16 xg[BATCH][KI];   // 6336 B
  __shared__ _Float16 Wn[KI * OG];     // 50688 B
  __shared__ float    biasn[OG];
  const int n = blockIdx.x;
  const int t = threadIdx.x;
  if (t < DDIM) En[t] = E[n * DDIM + t];
  __syncthreads();

  for (int idx = t; idx < BATCH * KI; idx += 256) {
    const int b = idx / KI, ki = idx % KI;
    const int k = ki / CIN, c = ki % CIN;
    const _Float16* src = (k == 0) ? X0T : (k == 1 ? X1T : X2T);
    xg[b][ki] = src[(size_t)(b * CIN + c) * NNODE + n];
  }
  for (int idx = t; idx < KI * OG; idx += 256) {
    const int ki = idx / OG, o = idx % OG;
    float a = 0.f;
#pragma unroll
    for (int d = 0; d < DDIM; ++d) a += En[d] * Wpool[((size_t)d * KI + ki) * OG + o];
    Wn[idx] = (_Float16)a;
  }
  if (t < OG) {
    float a = 0.f;
#pragma unroll
    for (int d = 0; d < DDIM; ++d) a += En[d] * Bpool[d * OG + t];
    biasn[t] = a;
  }
  __syncthreads();

  const int o = t & (OG - 1);
  const int bg = t >> 7;  // 0..1, 8 batches each
  float acc[8];
#pragma unroll
  for (int i = 0; i < 8; ++i) acc[i] = biasn[o];
  for (int ki = 0; ki < KI; ++ki) {
    const float wv = (float)Wn[ki * OG + o];
#pragma unroll
    for (int i = 0; i < 8; ++i) acc[i] += (float)xg[bg * 8 + i][ki] * wv;
  }
#pragma unroll
  for (int i = 0; i < 8; ++i) {
    const int b = bg * 8 + i;
    const float s = 1.f / (1.f + expf(-acc[i]));
    if (o < HID) {
      zbuf[((size_t)n * BATCH + b) * HID + o] = s;  // z
    } else {
      const int h = o - HID;                        // r -> candidate input
      const float rs = s * state[((size_t)b * NNODE + n) * HID + h];
      XcT[(size_t)(b * CIN + 2 + h) * NNODE + n] = (_Float16)rs;
    }
  }
}

// ---------------------------------------------------------------------------
// 5) Update per-node kernel: h = tanh(...); out = z*state + (1-z)*h
// ---------------------------------------------------------------------------
__global__ __launch_bounds__(256) void update_node_kernel(
    const _Float16* __restrict__ X0T, const _Float16* __restrict__ X1T,
    const _Float16* __restrict__ X2T, const float* __restrict__ E,
    const float* __restrict__ Wpool, const float* __restrict__ Bpool,
    const float* __restrict__ state, const float* __restrict__ zbuf,
    float* __restrict__ out) {
  __shared__ float    En[DDIM];
  __shared__ _Float16 xg[BATCH][KI];   // 6336 B
  __shared__ _Float16 Wn[KI * OU];     // 25344 B
  __shared__ float    biasn[OU];
  const int n = blockIdx.x;
  const int t = threadIdx.x;
  if (t < DDIM) En[t] = E[n * DDIM + t];
  __syncthreads();

  for (int idx = t; idx < BATCH * KI; idx += 256) {
    const int b = idx / KI, ki = idx % KI;
    const int k = ki / CIN, c = ki % CIN;
    const _Float16* src = (k == 0) ? X0T : (k == 1 ? X1T : X2T);
    xg[b][ki] = src[(size_t)(b * CIN + c) * NNODE + n];
  }
  for (int idx = t; idx < KI * OU; idx += 256) {
    const int ki = idx / OU, o = idx % OU;
    float a = 0.f;
#pragma unroll
    for (int d = 0; d < DDIM; ++d) a += En[d] * Wpool[((size_t)d * KI + ki) * OU + o];
    Wn[idx] = (_Float16)a;
  }
  if (t < OU) {
    float a = 0.f;
#pragma unroll
    for (int d = 0; d < DDIM; ++d) a += En[d] * Bpool[d * OU + t];
    biasn[t] = a;
  }
  __syncthreads();

  const int o = t & (OU - 1);
  const int bg = t >> 6;  // 0..3, 4 batches each
  float acc[4];
#pragma unroll
  for (int i = 0; i < 4; ++i) acc[i] = biasn[o];
  for (int ki = 0; ki < KI; ++ki) {
    const float wv = (float)Wn[ki * OU + o];
#pragma unroll
    for (int i = 0; i < 4; ++i) acc[i] += (float)xg[bg * 4 + i][ki] * wv;
  }
#pragma unroll
  for (int i = 0; i < 4; ++i) {
    const int b = bg * 4 + i;
    const float h  = tanhf(acc[i]);
    const float z  = zbuf[((size_t)n * BATCH + b) * HID + o];
    const float st = state[((size_t)b * NNODE + n) * HID + o];
    out[((size_t)b * NNODE + n) * HID + o] = z * st + (1.f - z) * h;
  }
}

// ---------------------------------------------------------------------------
extern "C" void kernel_launch(void* const* d_in, const int* in_sizes, int n_in,
                              void* d_out, int out_size, void* d_ws, size_t ws_size,
                              hipStream_t stream) {
  const float* x     = (const float*)d_in[0];
  const float* state = (const float*)d_in[1];
  const float* E     = (const float*)d_in[2];
  const float* gW    = (const float*)d_in[3];
  const float* gB    = (const float*)d_in[4];
  const float* uW    = (const float*)d_in[5];
  const float* uB    = (const float*)d_in[6];
  float* out = (float*)d_out;

  char* ws = (char*)d_ws;
  _Float16* A16 = (_Float16*)(ws);                    //  8,388,608 B
  _Float16* X0T = (_Float16*)(ws + 8388608);          //  4,325,376 B
  _Float16* X1T = (_Float16*)(ws + 12713984);         //  4,325,376 B
  _Float16* X2T = (_Float16*)(ws + 17039360);         //  4,325,376 B
  float*    zbf = (float*)   (ws + 21364736);         //  8,388,608 B (~28.4 MB total)

  build_A_kernel<<<NNODE, 256, 0, stream>>>(E, A16);
  pack_inp_kernel<<<(FDIM * NNODE + 255) / 256, 256, 0, stream>>>(x, state, X0T);

  dim3 gg(FDIM / 32, NNODE / 256);  // (33, 8)
  // Gate pass: X1 = A@X0 ; X2 = 2*A@X1 - X0
  gemm_prop_kernel<<<gg, 128, 0, stream>>>(A16, X0T, X0T, X1T, 1.f, 0.f);
  gemm_prop_kernel<<<gg, 128, 0, stream>>>(A16, X1T, X0T, X2T, 2.f, -1.f);
  gate_node_kernel<<<NNODE, 256, 0, stream>>>(X0T, X1T, X2T, E, gW, gB, state, zbf, X0T);
  // Candidate pass (X0T now holds concat(x, r*state)):
  gemm_prop_kernel<<<gg, 128, 0, stream>>>(A16, X0T, X0T, X1T, 1.f, 0.f);
  gemm_prop_kernel<<<gg, 128, 0, stream>>>(A16, X1T, X0T, X2T, 2.f, -1.f);
  update_node_kernel<<<NNODE, 256, 0, stream>>>(X0T, X1T, X2T, E, uW, uB, state, zbf, out);
}
